// PyramidCrossAttention_39092792328734
// MI455X (gfx1250) — compile-verified, matrix operand order checked by reading, not run on hardware
//
#include <hip/hip_runtime.h>

// ---------------------------------------------------------------------------
// PyramidCrossAttention for MI455X (gfx1250, wave32, WMMA bf16 16x16x32).
// All GEMM-shaped work (incl. grouped 3x3 conv as 9-tap implicit GEMM) on the
// WMMA pipe; weights pre-converted to bf16; activations stored transposed
// bf16 [N,K] so every fragment is two 16-byte global_load_b128s.
// Scratch requirement: ~130 MB in d_ws (layout in kernel_launch).
// ---------------------------------------------------------------------------

#define Bsz  8
#define Csz  128
#define HH   64
#define WW   64
#define HWsz 4096
#define NHsz 4
#define Lsz  3
#define Psz  4
#define DHsz 32

typedef __attribute__((ext_vector_type(16))) __bf16        v16bf;
typedef __attribute__((ext_vector_type(8)))  float         v8f;
typedef __attribute__((ext_vector_type(4)))  unsigned int  v4u;

union fragu { v16bf v; v4u q[2]; };

// ----------------------------- bf16 helpers --------------------------------
__device__ __forceinline__ unsigned short f2bf_bits(float f) {
  __bf16 b = (__bf16)f;                       // hardware cvt (RNE)
  unsigned short h; __builtin_memcpy(&h, &b, 2);
  return h;
}
__device__ __forceinline__ __bf16 f2bf(float f) { return (__bf16)f; }
__device__ __forceinline__ float bf2f(unsigned short h) {
  unsigned u = ((unsigned)h) << 16;
  float f; __builtin_memcpy(&f, &u, 4);
  return f;
}

// ------------------------- WMMA fragment loaders ---------------------------
// bf16 16x32 fragment layout (ISA 7.12.2), symmetric for A and B:
//  A: lane m=lane%16 reads its row;  B: lane n=lane%16 reads its column.
//  Elements 0..7  = k-offsets (kh+0 .. kh+7)   -> one 16B load
//  Elements 8..15 = k-offsets (kh+16 .. kh+23) -> one 16B load
//  kh = 8 for lanes 16..31.
__device__ __forceinline__ v16bf load_frag16(const unsigned short* __restrict__ rc,
                                             int kbase, int kh) {
  fragu f;
  f.q[0] = *reinterpret_cast<const v4u*>(rc + kbase + kh);
  f.q[1] = *reinterpret_cast<const v4u*>(rc + kbase + kh + 16);
  return f.v;
}
__device__ __forceinline__ v16bf zero_frag() {
  fragu f;
  f.q[0] = (v4u){0u, 0u, 0u, 0u};
  f.q[1] = (v4u){0u, 0u, 0u, 0u};
  return f.v;
}

// B fragment from fp32 [K,N] activations with per-k additive vector (valproj).
__device__ __forceinline__ v16bf load_bfrag_add(const float* __restrict__ X,
                                                const float* __restrict__ add, int ldn,
                                                int kbase, int nbase, int lane) {
  int kh = ((lane >> 4) & 1) << 3;
  int n = nbase + (lane & 15);
  v16bf b;
#pragma unroll
  for (int j = 0; j < 8; ++j) {
    int k0 = ((j < 4) ? 2 * j : 16 + 2 * (j - 4)) + kbase + kh;
    b[2 * j]     = f2bf(X[(size_t)k0 * ldn + n] + add[k0]);
    b[2 * j + 1] = f2bf(X[(size_t)(k0 + 1) * ldn + n] + add[k0 + 1]);
  }
  return b;
}

// --------------------------- generic WMMA GEMM -----------------------------
// Out[b][m][n] = sum_k Wt[m][k]*X[b][n][k] + bias[m] (+ (resA-resB)*hf).
// X: bf16 transposed [batch][N][K]; Wt: bf16 [M,K].
// grid = (N/64, nBatch), block = 128 (4 waves, one 16-col tile each).
template <int NMT, int NKT>
__global__ void gemm_bb(const unsigned short* __restrict__ X,
                        const unsigned short* __restrict__ Wt,
                        const float* __restrict__ bias, float* __restrict__ Out,
                        int N, long long sX, long long sO,
                        const float* __restrict__ resA, const float* __restrict__ resB,
                        const float* __restrict__ hfp) {
  constexpr int K = NKT * 32;
  const int lane  = threadIdx.x & 31;
  const int wave  = threadIdx.x >> 5;
  const int nbase = blockIdx.x * 64 + wave * 16;
  const int kh    = ((lane >> 4) & 1) << 3;
  const unsigned short* Xb   = X + (long long)blockIdx.y * sX;
  const unsigned short* bcol = Xb + (size_t)(nbase + (lane & 15)) * K;
  float* Ob = Out + (long long)blockIdx.y * sO;

  v8f acc[NMT];
#pragma unroll
  for (int i = 0; i < NMT; ++i) acc[i] = (v8f){};

#pragma unroll
  for (int kt = 0; kt < NKT; ++kt) {
    v16bf bf = load_frag16(bcol, kt * 32, kh);
#pragma unroll
    for (int mt = 0; mt < NMT; ++mt) {
      const unsigned short* arow = Wt + (size_t)(mt * 16 + (lane & 15)) * K;
      v16bf af = load_frag16(arow, kt * 32, kh);
      acc[mt] = __builtin_amdgcn_wmma_f32_16x16x32_bf16(false, af, false, bf,
                                                        (short)0, acc[mt], false, false);
    }
  }

  const int n    = nbase + (lane & 15);
  const int moff = ((lane >> 4) & 1) << 3;
  const float hs = resA ? hfp[0] : 0.f;
  const float* rA = resA ? resA + (long long)blockIdx.y * sO : nullptr;
  const float* rB = resB ? resB + (long long)blockIdx.y * sO : nullptr;
#pragma unroll
  for (int mt = 0; mt < NMT; ++mt) {
#pragma unroll
    for (int r = 0; r < 8; ++r) {
      int m = mt * 16 + moff + r;                 // C/D layout: vgpr r -> M=r(+8)
      size_t idx = (size_t)m * N + n;
      float v = acc[mt][r] + bias[m];
      if (rA) v += (rA[idx] - rB[idx]) * hs;
      Ob[idx] = v;
    }
  }
}

// ------- grouped 3x3 conv as 9-tap implicit WMMA GEMM, fused * wgate -------
// qbf: bf16 [B][HW][128]; wc: bf16 [128co][9tap][64ci].
// grid = (HW/64, B*2groups), block = 128.  Writes q_low fp32 [B,C,HW]
// (residual path) and qlow_bf bf16 [B,HW,C] (GEMM input path).
__global__ void convgemm_kernel(const unsigned short* __restrict__ qbf,
                                const unsigned short* __restrict__ wc,
                                const float* __restrict__ kb,
                                const float* __restrict__ wgate,
                                float* __restrict__ qlow,
                                unsigned short* __restrict__ qlow_bf) {
  const int lane  = threadIdx.x & 31;
  const int wave  = threadIdx.x >> 5;
  const int nbase = blockIdx.x * 64 + wave * 16;
  const int kh    = ((lane >> 4) & 1) << 3;
  const int b     = blockIdx.y >> 1;
  const int g     = blockIdx.y & 1;
  const int p     = nbase + (lane & 15);       // this lane's pixel (B-column)
  const int x = p & 63, y = p >> 6;

  v8f acc[4];
#pragma unroll
  for (int i = 0; i < 4; ++i) acc[i] = (v8f){};

  const unsigned short* qb = qbf + (size_t)b * HWsz * Csz + g * 64;

#pragma unroll
  for (int tap = 0; tap < 9; ++tap) {
    const int dy = tap / 3 - 1, dx = tap % 3 - 1;
    const bool valid = ((unsigned)(x + dx) < WW) && ((unsigned)(y + dy) < HH);
    const unsigned short* bcol = qb + (size_t)(p + dy * WW + dx) * Csz;
#pragma unroll
    for (int kt = 0; kt < 2; ++kt) {
      v16bf bf = valid ? load_frag16(bcol, kt * 32, kh) : zero_frag();
#pragma unroll
      for (int mt = 0; mt < 4; ++mt) {
        const unsigned short* arow =
            wc + ((size_t)(g * 64 + mt * 16 + (lane & 15)) * 9 + tap) * 64;
        v16bf af = load_frag16(arow, kt * 32, kh);
        acc[mt] = __builtin_amdgcn_wmma_f32_16x16x32_bf16(false, af, false, bf,
                                                          (short)0, acc[mt], false, false);
      }
    }
  }

  const int moff = ((lane >> 4) & 1) << 3;
#pragma unroll
  for (int mt = 0; mt < 4; ++mt) {
#pragma unroll
    for (int r = 0; r < 8; ++r) {
      int m = g * 64 + mt * 16 + moff + r;     // global output channel
      float v = (acc[mt][r] + kb[m]) * wgate[b * Csz + m];
      qlow[((size_t)b * Csz + m) * HWsz + p] = v;
      qlow_bf[((size_t)b * HWsz + p) * Csz + m] = f2bf_bits(v);
    }
  }
}

// -------- value projection: v_s[b*L+l][q][c] = bf16(valw @ (values+lev)) ----
__global__ void valproj_kernel(const float* __restrict__ values,
                               const unsigned short* __restrict__ valw_bf,
                               const float* __restrict__ valb, const float* __restrict__ lev,
                               unsigned short* __restrict__ Vs) {
  const int lane  = threadIdx.x & 31;
  const int wave  = threadIdx.x >> 5;
  const int nbase = blockIdx.x * 64 + wave * 16;
  const int kh    = ((lane >> 4) & 1) << 3;
  const int bl    = blockIdx.y;                  // b*L + l (values is [B,L,C,H,W])
  const float* Xb = values + (size_t)bl * Csz * HWsz;
  const float* lv = lev + (bl % Lsz) * Csz;

  v8f acc[8];
#pragma unroll
  for (int i = 0; i < 8; ++i) acc[i] = (v8f){};

#pragma unroll
  for (int kt = 0; kt < 4; ++kt) {
    v16bf bf = load_bfrag_add(Xb, lv, HWsz, kt * 32, nbase, lane);
#pragma unroll
    for (int mt = 0; mt < 8; ++mt) {
      const unsigned short* arow = valw_bf + (size_t)(mt * 16 + (lane & 15)) * Csz;
      v16bf af = load_frag16(arow, kt * 32, kh);
      acc[mt] = __builtin_amdgcn_wmma_f32_16x16x32_bf16(false, af, false, bf,
                                                        (short)0, acc[mt], false, false);
    }
  }
  const int n    = nbase + (lane & 15);
  const int moff = ((lane >> 4) & 1) << 3;
  unsigned short* op = Vs + ((size_t)bl * HWsz + n) * Csz;   // pixel-major, c contiguous
#pragma unroll
  for (int mt = 0; mt < 8; ++mt)
#pragma unroll
    for (int r = 0; r < 8; ++r) {
      int c = mt * 16 + moff + r;
      op[c] = f2bf_bits(acc[mt][r] + valb[c]);
    }
}

// ---------------------- fp32 -> bf16 weight conversion ---------------------
__global__ void cvt_bf16_kernel(const float* __restrict__ src,
                                unsigned short* __restrict__ dst, int n) {
  int i = blockIdx.x * 256 + threadIdx.x;
  if (i < n) dst[i] = f2bf_bits(src[i]);
}

// fd_k [co][ci][3][3] fp32 -> wc [co][tap][ci] bf16 (K=ci contiguous rows)
__global__ void packconv_kernel(const float* __restrict__ fdk,
                                unsigned short* __restrict__ wc) {
  int i = blockIdx.x * 256 + threadIdx.x;        // 128*9*64 = 73728
  if (i >= 128 * 9 * 64) return;
  int ci  = i & 63;
  int tap = (i >> 6) % 9;
  int co  = i / (9 * 64);
  wc[i] = f2bf_bits(fdk[((size_t)co * 64 + ci) * 9 + tap]);
}

// query [B,C,HW] fp32 -> query_bf [B,HW,C] bf16
__global__ void qtrans_kernel(const float* __restrict__ q,
                              unsigned short* __restrict__ qbf) {
  size_t idx = (size_t)blockIdx.x * blockDim.x + threadIdx.x;
  if (idx >= (size_t)Bsz * Csz * HWsz) return;
  int pix = (int)(idx & (HWsz - 1));
  int c   = (int)(idx >> 12) & 127;
  int b   = (int)(idx >> 19);
  qbf[((size_t)b * HWsz + pix) * Csz + c] = f2bf_bits(q[idx]);
}

// ------------------- pooled mean + gate MLP (wgate) ------------------------
__global__ void gates_kernel(const float* __restrict__ query,
                             const float* __restrict__ w1, const float* __restrict__ b1,
                             const float* __restrict__ w2, const float* __restrict__ b2,
                             float* __restrict__ wgate) {
  __shared__ float pooled[Csz];
  __shared__ float hg[32];
  const int b = blockIdx.x, t = threadIdx.x;
  if (t < Csz) pooled[t] = 0.f;
  __syncthreads();
  const float* qb = query + (size_t)b * Csz * HWsz;
  int c = t >> 1, half = t & 1;
  {
    const float* p = qb + (size_t)c * HWsz + half * (HWsz / 2);
    float s = 0.f;
    for (int i = 0; i < HWsz / 2; ++i) s += p[i];
    atomicAdd(&pooled[c], s);
  }
  __syncthreads();
  if (t < Csz) pooled[t] *= (1.f / (float)HWsz);
  __syncthreads();
  if (t < 32) {
    float s = b1[t];
    for (int k = 0; k < Csz; ++k) s += w1[t * Csz + k] * pooled[k];
    hg[t] = fmaxf(s, 0.f);
  }
  __syncthreads();
  if (t < Csz) {
    float s = b2[t];
    for (int k = 0; k < 32; ++k) s += w2[t * 32 + k] * hg[k];
    wgate[b * Csz + t] = 1.f / (1.f + __expf(-s));
  }
}

// ------------------- GroupNorm stats over u[2][B][64][HW] ------------------
__global__ void gnstats_kernel(const float* __restrict__ u, float* __restrict__ stats) {
  __shared__ float ssum[256], ssq[256];
  const int blk = blockIdx.x;                 // br*64 + b*8 + g
  const int g = blk & 7, b = (blk >> 3) & 7, br = blk >> 6;
  const float* p = u + (((size_t)br * Bsz + b) * 64 + g * 8) * HWsz;
  float s = 0.f, q = 0.f;
  for (int i = threadIdx.x; i < 8 * HWsz; i += 256) { float v = p[i]; s += v; q += v * v; }
  ssum[threadIdx.x] = s; ssq[threadIdx.x] = q;
  __syncthreads();
  for (int off = 128; off > 0; off >>= 1) {
    if (threadIdx.x < off) { ssum[threadIdx.x] += ssum[threadIdx.x + off];
                             ssq[threadIdx.x]  += ssq[threadIdx.x + off]; }
    __syncthreads();
  }
  if (threadIdx.x == 0) {
    float inv = 1.f / (8.f * HWsz);
    float mean = ssum[0] * inv;
    float var  = ssq[0] * inv - mean * mean;
    stats[blk * 2 + 0] = mean;
    stats[blk * 2 + 1] = rsqrtf(var + 1e-5f);
  }
}

// Normalize + ReLU; writes bf16 transposed [2B][HW][64] for WMMA consumption.
__global__ void gnnorm_kernel(const float* __restrict__ u, const float* __restrict__ stats,
                              const float* __restrict__ og, const float* __restrict__ ob,
                              const float* __restrict__ ag, const float* __restrict__ ab,
                              unsigned short* __restrict__ u_bf) {
  size_t idx = (size_t)blockIdx.x * blockDim.x + threadIdx.x;
  if (idx >= (size_t)2 * Bsz * 64 * HWsz) return;
  int pix = (int)(idx & (HWsz - 1));
  int ch = (int)(idx >> 12) & 63;
  int b  = (int)(idx >> 18) & 7;
  int br = (int)(idx >> 21);
  int sidx = (br * Bsz + b) * 8 + (ch >> 3);
  float mean = stats[sidx * 2], rstd = stats[sidx * 2 + 1];
  const float* gam = br ? ag : og;
  const float* bet = br ? ab : ob;
  float v = fmaxf((u[idx] - mean) * rstd * gam[ch] + bet[ch], 0.f);
  u_bf[(((size_t)br * Bsz + b) * HWsz + pix) * 64 + ch] = f2bf_bits(v);
}

// -------------- sampling locations + softmax attention weights -------------
__global__ void locattn_kernel(const float* __restrict__ offb, const float* __restrict__ awb,
                               float* __restrict__ locs, float* __restrict__ attw) {
  int idx = blockIdx.x * 256 + threadIdx.x;          // ((b*HW+q)*NH+nh)
  if (idx >= Bsz * HWsz * NHsz) return;
  int nh = idx & 3;
  int q  = (idx >> 2) & (HWsz - 1);
  int b  = idx >> 14;
  int x = q & 63, y = q >> 6;

  const float* ap = awb + ((size_t)b * 48 + nh * 12) * HWsz + q;
  float a[12], mx = -1e30f;
#pragma unroll
  for (int lp = 0; lp < 12; ++lp) { a[lp] = ap[(size_t)lp * HWsz]; mx = fmaxf(mx, a[lp]); }
  float sum = 0.f;
#pragma unroll
  for (int lp = 0; lp < 12; ++lp) { a[lp] = __expf(a[lp] - mx); sum += a[lp]; }
  float inv = 1.f / sum;
  float* wout = attw + (size_t)idx * 12;
#pragma unroll
  for (int lp = 0; lp < 12; ++lp) wout[lp] = a[lp] * inv;

  float rx = (float)x * (1.f / 63.f);                // linspace(0,1,64)
  float ry = (float)y * (1.f / 63.f);
  const float* op = offb + ((size_t)b * 96 + nh * 24) * HWsz + q;
  float* lout = locs + (size_t)idx * 24;
#pragma unroll
  for (int lp = 0; lp < 12; ++lp) {
    float ox = op[(size_t)(lp * 2 + 0) * HWsz] * (1.f / WW);
    float oy = op[(size_t)(lp * 2 + 1) * HWsz] * (1.f / HH);
    lout[lp * 2 + 0] = fminf(fmaxf(rx + ox, 0.f), 1.f);
    lout[lp * 2 + 1] = fminf(fmaxf(ry + oy, 0.f), 1.f);
  }
}

// -------- bilinear gather + attention combine: wave per (b,q,head) ---------
// Output bf16 transposed [B][HW][C] (coalesced store, feeds final WMMA GEMM).
__global__ void sample_kernel(const unsigned short* __restrict__ Vs,
                              const float* __restrict__ locs, const float* __restrict__ attw,
                              unsigned short* __restrict__ hout_bf) {
  const int lane = threadIdx.x & 31;                 // dh
  const int nh   = threadIdx.x >> 5;                 // 4 waves = 4 heads
  const int bq   = blockIdx.x;                       // b*HW + q
  const int b    = bq >> 12;
  const int l0   = b * Lsz;
  const size_t base = (size_t)bq * NHsz + nh;
  const float* lp = locs + base * 24;
  const float* wp = attw + base * 12;

  float acc = 0.f;
  for (int lpi = 0; lpi < 12; ++lpi) {
    int l = lpi >> 2;
    float w  = wp[lpi];
    float xf = lp[lpi * 2 + 0] * WW - 0.5f;
    float yf = lp[lpi * 2 + 1] * HH - 0.5f;
    float x0f = floorf(xf), y0f = floorf(yf);
    float wx1 = xf - x0f, wy1 = yf - y0f;
    int x0 = (int)x0f, y0 = (int)y0f;
    const unsigned short* vb =
        Vs + ((size_t)(l0 + l) * HWsz) * Csz + nh * DHsz + lane;
#pragma unroll
    for (int cy = 0; cy < 2; ++cy) {
#pragma unroll
      for (int cx = 0; cx < 2; ++cx) {
        int xi = x0 + cx, yi = y0 + cy;
        if (xi >= 0 && xi < WW && yi >= 0 && yi < HH) {
          float cw = (cx ? wx1 : 1.f - wx1) * (cy ? wy1 : 1.f - wy1);
          acc += w * cw * bf2f(vb[(size_t)(yi * WW + xi) * Csz]);
        }
      }
    }
  }
  hout_bf[(size_t)bq * Csz + nh * DHsz + lane] = f2bf_bits(acc);
}

// ---------------------------------------------------------------------------
extern "C" void kernel_launch(void* const* d_in, const int* in_sizes, int n_in,
                              void* d_out, int out_size, void* d_ws, size_t ws_size,
                              hipStream_t stream) {
  (void)in_sizes; (void)n_in; (void)out_size; (void)ws_size;
  const float* query  = (const float*)d_in[0];
  // d_in[1] = keys (unused by reference)
  const float* values = (const float*)d_in[2];
  const float* fd_w1  = (const float*)d_in[3];
  const float* fd_b1  = (const float*)d_in[4];
  const float* fd_w2  = (const float*)d_in[5];
  const float* fd_b2  = (const float*)d_in[6];
  const float* fd_k   = (const float*)d_in[7];
  const float* fd_kb  = (const float*)d_in[8];
  const float* off_w1 = (const float*)d_in[9];
  const float* off_b1 = (const float*)d_in[10];
  const float* off_g  = (const float*)d_in[11];
  const float* off_be = (const float*)d_in[12];
  const float* off_w2 = (const float*)d_in[13];
  const float* off_b2 = (const float*)d_in[14];
  const float* aw_w1  = (const float*)d_in[15];
  const float* aw_b1  = (const float*)d_in[16];
  const float* aw_g   = (const float*)d_in[17];
  const float* aw_be  = (const float*)d_in[18];
  const float* aw_w2  = (const float*)d_in[19];
  const float* aw_b2  = (const float*)d_in[20];
  const float* val_w  = (const float*)d_in[21];
  const float* val_b  = (const float*)d_in[22];
  const float* out_w  = (const float*)d_in[23];
  const float* out_b  = (const float*)d_in[24];
  const float* lev    = (const float*)d_in[25];
  const float* hf     = (const float*)d_in[26];
  float* out = (float*)d_out;

  // ---- scratch layout, total ~130 MB ----
  // bf16 weight arena first (16B aligned; sub-offsets multiples of 8 elems)
  unsigned short* wbf = (unsigned short*)d_ws;        // 58368 + 73728 ushorts
  unsigned short* woff1_bf = wbf;                     // 64*128   = 8192
  unsigned short* waw1_bf  = wbf + 8192;              // 64*128   = 8192
  unsigned short* woff2_bf = wbf + 16384;             // 96*64    = 6144
  unsigned short* waw2_bf  = wbf + 22528;             // 48*64    = 3072
  unsigned short* wval_bf  = wbf + 25600;             // 128*128  = 16384
  unsigned short* wout_bf  = wbf + 41984;             // 128*128  = 16384
  unsigned short* wconv_bf = wbf + 58368;             // 128*9*64 = 73728
  float* fws     = (float*)(wbf + 132096);
  float* wgate   = fws;                               // 1024
  float* q_low   = wgate   + 1024;                    // B*C*HW     = 4194304
  float* u       = q_low   + 4194304;                 // 2*B*64*HW  = 4194304
  float* stats   = u       + 4194304;                 // 256
  float* off_out = stats   + 256;                     // B*96*HW    = 3145728
  float* aw_out  = off_out + 3145728;                 // B*48*HW    = 1572864
  float* locs    = aw_out  + 1572864;                 // B*HW*NH*24 = 3145728
  float* attw    = locs    + 3145728;                 // B*HW*NH*12 = 1572864
  unsigned short* query_bf = (unsigned short*)(attw + 1572864);      // B*HW*128
  unsigned short* qlow_bf  = query_bf + (size_t)Bsz * HWsz * Csz;    // B*HW*128
  unsigned short* u_bf     = qlow_bf + (size_t)Bsz * HWsz * Csz;     // 2B*HW*64
  unsigned short* hout_bf  = u_bf + (size_t)2 * Bsz * HWsz * 64;     // B*HW*128
  unsigned short* vs       = hout_bf + (size_t)Bsz * HWsz * Csz;     // B*L*HW*C

  const long long sQ   = (long long)Csz * HWsz;   // per-batch stride, [B,C,HW] fp32
  const long long sXq  = (long long)HWsz * Csz;   // per-batch stride, [B,HW,C] bf16
  const long long sXu  = (long long)HWsz * 64;    // per-batch stride, [*,HW,64] bf16
  float* u_off = u;
  float* u_aw  = u + (size_t)Bsz * 64 * HWsz;
  unsigned short* u_bf_off = u_bf;
  unsigned short* u_bf_aw  = u_bf + (size_t)Bsz * HWsz * 64;

  // 0) weights -> bf16 (once per launch; tiny)
  cvt_bf16_kernel<<<32, 256, 0, stream>>>(off_w1, woff1_bf, 8192);
  cvt_bf16_kernel<<<32, 256, 0, stream>>>(aw_w1,  waw1_bf,  8192);
  cvt_bf16_kernel<<<24, 256, 0, stream>>>(off_w2, woff2_bf, 6144);
  cvt_bf16_kernel<<<12, 256, 0, stream>>>(aw_w2,  waw2_bf,  3072);
  cvt_bf16_kernel<<<64, 256, 0, stream>>>(val_w,  wval_bf,  16384);
  cvt_bf16_kernel<<<64, 256, 0, stream>>>(out_w,  wout_bf,  16384);
  packconv_kernel<<<(128 * 9 * 64 + 255) / 256, 256, 0, stream>>>(fd_k, wconv_bf);
  qtrans_kernel<<<(Bsz * Csz * HWsz + 255) / 256, 256, 0, stream>>>(query, query_bf);

  // 1) gates
  gates_kernel<<<Bsz, 256, 0, stream>>>(query, fd_w1, fd_b1, fd_w2, fd_b2, wgate);
  // 2) grouped 3x3 conv (implicit WMMA GEMM) * wgate -> q_low + qlow_bf
  convgemm_kernel<<<dim3(HWsz / 64, Bsz * 2), 128, 0, stream>>>(
      query_bf, wconv_bf, fd_kb, wgate, q_low, qlow_bf);
  // 3) first conv1x1s (WMMA): qlow_bf -> u_off / u_aw fp32 (M=64,K=128)
  gemm_bb<4, 4><<<dim3(HWsz / 64, Bsz), 128, 0, stream>>>(
      qlow_bf, woff1_bf, off_b1, u_off, HWsz, sXq, (long long)64 * HWsz,
      nullptr, nullptr, nullptr);
  gemm_bb<4, 4><<<dim3(HWsz / 64, Bsz), 128, 0, stream>>>(
      qlow_bf, waw1_bf, aw_b1, u_aw, HWsz, sXq, (long long)64 * HWsz,
      nullptr, nullptr, nullptr);
  // 4) GroupNorm stats + normalize/ReLU -> u_bf (bf16 [2B,HW,64])
  gnstats_kernel<<<2 * Bsz * 8, 256, 0, stream>>>(u, stats);
  gnnorm_kernel<<<(2 * Bsz * 64 * HWsz + 255) / 256, 256, 0, stream>>>(
      u, stats, off_g, off_be, aw_g, aw_be, u_bf);
  // 5) second conv1x1s (WMMA): M=96,K=64 and M=48,K=64
  gemm_bb<6, 2><<<dim3(HWsz / 64, Bsz), 128, 0, stream>>>(
      u_bf_off, woff2_bf, off_b2, off_out, HWsz, sXu, (long long)96 * HWsz,
      nullptr, nullptr, nullptr);
  gemm_bb<3, 2><<<dim3(HWsz / 64, Bsz), 128, 0, stream>>>(
      u_bf_aw, waw2_bf, aw_b2, aw_out, HWsz, sXu, (long long)48 * HWsz,
      nullptr, nullptr, nullptr);
  // 6) locations + softmax
  locattn_kernel<<<(Bsz * HWsz * NHsz + 255) / 256, 256, 0, stream>>>(
      off_out, aw_out, locs, attw);
  // 7) value projection (WMMA, +level_embed, bf16 pixel-major store)
  valproj_kernel<<<dim3(HWsz / 64, Bsz * Lsz), 128, 0, stream>>>(
      values, wval_bf, val_b, lev, vs);
  // 8) bilinear sampling + attention combine -> hout_bf (bf16 [B,HW,C])
  sample_kernel<<<Bsz * HWsz, 128, 0, stream>>>(vs, locs, attw, hout_bf);
  // 9) output projection (WMMA) + (query - q_low) * hf_scale residual
  gemm_bb<8, 4><<<dim3(HWsz / 64, Bsz), 128, 0, stream>>>(
      hout_bf, wout_bf, out_b, out, HWsz, sXq, sQ, query, q_low, hf);
}
